// SparseMoE_73065983640086
// MI455X (gfx1250) — matrix operand order repaired and checked by reading, not compiled, not run on hardware
//
#include <hip/hip_runtime.h>
#include <hip/hip_bf16.h>

#define B_   4
#define S_   2048
#define D_   1024
#define H_   2048
#define E_   8
#define KTOP 2
#define NT   (B_ * S_)          // 8192 tokens
#define TILES_PER_E 512         // worst case NT/16 tiles per expert

typedef __attribute__((ext_vector_type(16))) __bf16 v16bf;
typedef __attribute__((ext_vector_type(8)))  float  v8f;

__device__ __forceinline__ unsigned short f2bf(float f) {
    unsigned int u = __float_as_uint(f);
    u += 0x7fffu + ((u >> 16) & 1u);          // round-to-nearest-even
    return (unsigned short)(u >> 16);
}

// ---------------------------------------------------------------- zero / init
__global__ void SparseMoE_zero_kernel(float* __restrict__ fout,
                                      int* __restrict__ ecnt,
                                      float* __restrict__ usage) {
    size_t i = (size_t)blockIdx.x * blockDim.x + threadIdx.x;
    size_t n = (size_t)NT * D_;
    size_t stride = (size_t)gridDim.x * blockDim.x;
    for (size_t j = i; j < n; j += stride) fout[j] = 0.0f;
    if (i < E_) { ecnt[i] = 0; usage[i] = 0.0f; }
}

// ------------------------------------------- repack w1 -> bf16 B-frag layout
// wb1 layout: [e][hb(128)][kb(32)][lane(32)][t(16)]  (B frag: lane=N+16*((K>>3)&1), t=(K&7)+8*((K>>4)&1))
__global__ void SparseMoE_prep_wb1(const float* __restrict__ w1,
                                   unsigned short* __restrict__ wb1) {
    unsigned int stride = gridDim.x * blockDim.x;
    for (unsigned int i = blockIdx.x * blockDim.x + threadIdx.x;
         i < (unsigned int)(E_ * D_ * H_); i += stride) {
        unsigned int t  = i & 15u;
        unsigned int ln = (i >> 4) & 31u;
        unsigned int kb = (i >> 9) & 31u;
        unsigned int hb = (i >> 14) & 127u;
        unsigned int e  = i >> 21;
        unsigned int d  = kb * 32u + ((ln >= 16u) ? 8u : 0u) + ((t >= 8u) ? 16u : 0u) + (t & 7u);   // K index
        unsigned int h  = hb * 16u + (ln & 15u);                     // N index
        wb1[i] = f2bf(w1[((size_t)e * D_ + d) * H_ + h]);
    }
}

// wb2 layout: [e][nb(64)][kb(64)][lane(32)][t(16)]
__global__ void SparseMoE_prep_wb2(const float* __restrict__ w2,
                                   unsigned short* __restrict__ wb2) {
    unsigned int stride = gridDim.x * blockDim.x;
    for (unsigned int i = blockIdx.x * blockDim.x + threadIdx.x;
         i < (unsigned int)(E_ * H_ * D_); i += stride) {
        unsigned int t  = i & 15u;
        unsigned int ln = (i >> 4) & 31u;
        unsigned int kb = (i >> 9) & 63u;
        unsigned int nb = (i >> 15) & 63u;
        unsigned int e  = i >> 21;
        unsigned int h  = kb * 32u + ((ln >= 16u) ? 8u : 0u) + ((t >= 8u) ? 16u : 0u) + (t & 7u);   // K index
        unsigned int d  = nb * 16u + (ln & 15u);                     // N index
        wb2[i] = f2bf(w2[((size_t)e * H_ + h) * D_ + d]);
    }
}

// ------------------------------------- spiking norm + gating + top-2 routing
__global__ __launch_bounds__(256)
void SparseMoE_gate_kernel(const float* __restrict__ x,
                           const float* __restrict__ gw,
                           const float* __restrict__ gb,
                           const float* __restrict__ noise,
                           unsigned short* __restrict__ xn,
                           float* __restrict__ out_topi,
                           int* __restrict__ ecnt,
                           float* __restrict__ usage,
                           int* __restrict__ etok,
                           float* __restrict__ ep) {
    int tok  = (int)((blockIdx.x * blockDim.x + threadIdx.x) >> 5);
    int lane = threadIdx.x & 31;
    if (tok >= NT) return;

    const float* xr = x + (size_t)tok * D_;
    float xv[32];
    float s = 0.0f;
#pragma unroll
    for (int i = 0; i < 32; ++i) { xv[i] = xr[lane + 32 * i]; s += xv[i]; }
#pragma unroll
    for (int o = 16; o > 0; o >>= 1) s += __shfl_xor(s, o, 32);

    float mean  = s * (1.0f / (float)D_);
    float keep  = (mean > 0.1f) ? 1.0f : 0.0f;
    float denom = keep * s + 1e-8f;
    float inv   = keep / denom;

    float lg[E_];
#pragma unroll
    for (int e = 0; e < E_; ++e) lg[e] = 0.0f;

#pragma unroll
    for (int i = 0; i < 32; ++i) {
        int d = lane + 32 * i;
        float v = xv[i] * inv;
        xn[(size_t)tok * D_ + d] = f2bf(v);
        const float* gr = gw + (size_t)d * E_;
#pragma unroll
        for (int e = 0; e < E_; ++e) lg[e] += v * gr[e];
    }
#pragma unroll
    for (int e = 0; e < E_; ++e) {
#pragma unroll
        for (int o = 16; o > 0; o >>= 1) lg[e] += __shfl_xor(lg[e], o, 32);
    }

    if (lane == 0) {
        float l[E_];
#pragma unroll
        for (int e = 0; e < E_; ++e)
            l[e] = lg[e] + gb[e] + noise[(size_t)tok * E_ + e] * 0.01f;
        int i1 = 0;
#pragma unroll
        for (int e = 1; e < E_; ++e) if (l[e] > l[i1]) i1 = e;
        int i2 = (i1 == 0) ? 1 : 0;
#pragma unroll
        for (int e = 0; e < E_; ++e) if (e != i1 && l[e] > l[i2]) i2 = e;
        float thr = l[i2];          // min of top-2
        float m   = l[i1];
        float p[E_]; float psum = 0.0f;
#pragma unroll
        for (int e = 0; e < E_; ++e) {
            float pe = (l[e] >= thr) ? __expf(l[e] - m) : 0.0f;
            p[e] = pe; psum += pe;
        }
        float rs = 1.0f / psum;
#pragma unroll
        for (int e = 0; e < E_; ++e) {
            p[e] *= rs;
            atomicAdd(&usage[e], p[e]);
        }
        out_topi[tok * 2 + 0] = (float)i1;
        out_topi[tok * 2 + 1] = (float)i2;
        int pos = atomicAdd(&ecnt[i1], 1);
        etok[i1 * NT + pos] = tok; ep[i1 * NT + pos] = p[i1];
        pos = atomicAdd(&ecnt[i2], 1);
        etok[i2 * NT + pos] = tok; ep[i2 * NT + pos] = p[i2];
    }
}

// ------------------------------------------------------------- aux loss
__global__ void SparseMoE_aux_kernel(const float* __restrict__ usage,
                                     float* __restrict__ out_aux) {
    if (threadIdx.x == 0 && blockIdx.x == 0) {
        float s = 0.0f;
        for (int e = 0; e < E_; ++e) s += usage[e];
        float imp[E_]; float m = 0.0f;
        for (int e = 0; e < E_; ++e) { imp[e] = usage[e] / (s + 1e-10f); m += imp[e]; }
        m *= (1.0f / E_);
        float var = 0.0f;
        for (int e = 0; e < E_; ++e) { float d = imp[e] - m; var += d * d; }
        var *= (1.0f / E_);
        out_aux[0] = sqrtf(var) / (m + 1e-10f);
    }
}

// ---------------------------------------------- routed fused FFN (WMMA bf16)
__global__ __launch_bounds__(256)
void SparseMoE_ffn_kernel(const unsigned short* __restrict__ xn,
                          const unsigned short* __restrict__ wb1,
                          const unsigned short* __restrict__ wb2,
                          const float* __restrict__ b1,
                          const float* __restrict__ b2,
                          const int* __restrict__ ecnt,
                          const int* __restrict__ etok,
                          const float* __restrict__ ep,
                          float* __restrict__ fout) {
    int e    = blockIdx.x / TILES_PER_E;
    int tile = blockIdx.x % TILES_PER_E;
    int cnt  = ecnt[e];
    int base = tile * 16;
    if (base >= cnt) return;

    __shared__ __align__(32) unsigned short xA[16384];  // 16x1024 bf16, A-frag layout
    __shared__ __align__(32) unsigned short hA[2048];   // 16x128  bf16, A-frag layout
    __shared__ int   rowTok[16];
    __shared__ float rowP[16];

    int tid = threadIdx.x, lane = tid & 31, wv = tid >> 5;

    if (tid < 16) {
        int idx = base + tid;
        if (idx < cnt) { rowTok[tid] = etok[e * NT + idx]; rowP[tid] = ep[e * NT + idx]; }
        else           { rowTok[tid] = etok[e * NT + base]; rowP[tid] = 0.0f; }
    }
    __syncthreads();

    // stage x rows into LDS pre-packed as 16x32 bf16 A fragments (ISA 7.12.2)
    for (int i = tid; i < 16384; i += 256) {
        int t  = i & 15;
        int ln = (i >> 4) & 31;
        int kb = i >> 9;
        int row = ln & 15;
        int k = kb * 32 + ((ln < 16) ? 0 : 8) + ((t < 8) ? 0 : 16)
              + 2 * ((t >> 1) & 3) + (t & 1);
        xA[i] = xn[(size_t)rowTok[row] * D_ + k];
    }
    __syncthreads();

    v8f acc[8];
#pragma unroll
    for (int j = 0; j < 8; ++j) acc[j] = (v8f){0,0,0,0,0,0,0,0};

    const v16bf* xAf = (const v16bf*)xA;   // frag element: [kb*32 + lane]
    const v16bf* hAf = (const v16bf*)hA;
    const v16bf* w1f = (const v16bf*)wb1;
    const v16bf* w2f = (const v16bf*)wb2;

    int Nc = lane & 15;

    for (int hc = 0; hc < 16; ++hc) {
        // ---- stage 1: h[16 x 16] for columns hb*16.. (wave wv), K = D = 1024
        int hb = hc * 8 + wv;
        const v16bf* bp1 = w1f + (((size_t)e * 128 + hb) * 32) * 32;
        v8f hacc = (v8f){0,0,0,0,0,0,0,0};
#pragma unroll 4
        for (int kb = 0; kb < 32; ++kb) {
            v16bf a = xAf[kb * 32 + lane];
            v16bf b = bp1[(size_t)kb * 32 + lane];
            if (kb + 8 < 32) __builtin_prefetch(&bp1[(size_t)(kb + 8) * 32 + lane], 0, 0);
            hacc = __builtin_amdgcn_wmma_f32_16x16x32_bf16(
                false, a, false, b, (short)0, hacc, false, false);
        }
        float bb = b1[e * H_ + hb * 16 + Nc];

        __syncthreads();                    // stage-2 readers of hA are done
        // bias + SiLU, repack into stage-2 A-frag layout
#pragma unroll
        for (int r = 0; r < 8; ++r) {
            float v = hacc[r] + bb;
            v = v / (1.0f + __expf(-v));
            int M  = r + ((lane < 16) ? 0 : 8);
            int Kr = wv * 16 + Nc;          // K within 128-chunk
            int kb2    = Kr >> 5;
            int kk     = Kr & 31;
            int lanehi = ((kk & 15) >= 8) ? 16 : 0;
            int hi16   = (kk >= 16) ? 8 : 0;
            int kk2    = kk - (lanehi ? 8 : 0) - (hi16 ? 16 : 0);   // 0..7
            int idx    = hi16 + kk2;                                // packed slot
            hA[kb2 * 512 + (M + lanehi) * 16 + idx] = f2bf(v);
        }
        __syncthreads();

        // ---- stage 2: out[16 x 128] strip of wave wv, K-chunk = 128 of H
#pragma unroll
        for (int j = 0; j < 8; ++j) {
            int nb = wv * 8 + j;
            const v16bf* bp2 = w2f + (((size_t)e * 64 + nb) * 64) * 32;
#pragma unroll
            for (int kbl = 0; kbl < 4; ++kbl) {
                int kb = hc * 4 + kbl;
                v16bf a = hAf[kbl * 32 + lane];
                v16bf b = bp2[(size_t)kb * 32 + lane];
                acc[j] = __builtin_amdgcn_wmma_f32_16x16x32_bf16(
                    false, a, false, b, (short)0, acc[j], false, false);
            }
        }
        __syncthreads();
    }

    // combine: final += p * (acc + b2)
#pragma unroll
    for (int j = 0; j < 8; ++j) {
        int col = wv * 128 + j * 16 + Nc;
        float b2v = b2[e * D_ + col];
#pragma unroll
        for (int r = 0; r < 8; ++r) {
            int M = r + ((lane < 16) ? 0 : 8);
            float val = rowP[M] * (acc[j][r] + b2v);
            atomicAdd(&fout[(size_t)rowTok[M] * D_ + col], val);
        }
    }
}

// ---------------------------------------------------------------- launcher
extern "C" void kernel_launch(void* const* d_in, const int* in_sizes, int n_in,
                              void* d_out, int out_size, void* d_ws, size_t ws_size,
                              hipStream_t stream) {
    const float* x     = (const float*)d_in[0];
    const float* gw    = (const float*)d_in[1];
    const float* gb    = (const float*)d_in[2];
    const float* w1    = (const float*)d_in[3];
    const float* b1    = (const float*)d_in[4];
    const float* w2    = (const float*)d_in[5];
    const float* b2    = (const float*)d_in[6];
    const float* noise = (const float*)d_in[7];

    float* fout  = (float*)d_out;                  // [NT, D]
    float* topi  = fout + (size_t)NT * D_;         // [NT, K] as float
    float* auxo  = topi + (size_t)NT * KTOP;       // scalar

    char* ws = (char*)d_ws;
    unsigned short* xn  = (unsigned short*)ws;                    ws += (size_t)NT * D_ * 2;       // 16 MB
    unsigned short* wb1 = (unsigned short*)ws;                    ws += (size_t)E_ * D_ * H_ * 2;  // 32 MB
    unsigned short* wb2 = (unsigned short*)ws;                    ws += (size_t)E_ * H_ * D_ * 2;  // 32 MB
    int*   ecnt  = (int*)ws;                                      ws += 256;
    float* usage = (float*)ws;                                    ws += 256;
    int*   etok  = (int*)ws;                                      ws += (size_t)E_ * NT * 4;
    float* epv   = (float*)ws;                                    ws += (size_t)E_ * NT * 4;

    hipLaunchKernelGGL(SparseMoE_zero_kernel, dim3(4096), dim3(256), 0, stream,
                       fout, ecnt, usage);
    hipLaunchKernelGGL(SparseMoE_prep_wb1, dim3(8192), dim3(256), 0, stream, w1, wb1);
    hipLaunchKernelGGL(SparseMoE_prep_wb2, dim3(8192), dim3(256), 0, stream, w2, wb2);
    hipLaunchKernelGGL(SparseMoE_gate_kernel, dim3(NT / 8), dim3(256), 0, stream,
                       x, gw, gb, noise, xn, topi, ecnt, usage, etok, epv);
    hipLaunchKernelGGL(SparseMoE_aux_kernel, dim3(1), dim3(32), 0, stream, usage, auxo);
    hipLaunchKernelGGL(SparseMoE_ffn_kernel, dim3(E_ * TILES_PER_E), dim3(256), 0, stream,
                       xn, wb1, wb2, b1, b2, ecnt, etok, epv, fout);
}
